// Attention_55018531062102
// MI455X (gfx1250) — matrix operand order repaired and checked by reading, not compiled, not run on hardware
//
#include <hip/hip_runtime.h>
#include <hip/hip_bf16.h>

// ---------------------------------------------------------------------------
// MHA for MI455X (gfx1250): fp32 -> bf16 once, then all GEMMs + flash
// attention via v_wmma_f32_16x16x32_bf16 (fp32 accumulate), with CDNA5
// async global->LDS copies (ASYNCcnt) and double-buffered LDS tiles.
// B=4, S=2048, E=1024, H=16, D=64.
// ---------------------------------------------------------------------------

typedef __attribute__((ext_vector_type(16))) __bf16 v16bf;
typedef __attribute__((ext_vector_type(8)))  float  v8f;
typedef __attribute__((ext_vector_type(4)))  int    v4i_;

union FragB { uint4 q[2]; v16bf v; };

__device__ __forceinline__ unsigned short f2bf(float f) {
  union { __bf16 h; unsigned short u; } cv;
  cv.h = (__bf16)f;
  return cv.u;
}

// ---- CDNA5 async global->LDS 16B copy (ASYNCcnt-tracked, no VGPR bounce) --
#if __has_builtin(__builtin_amdgcn_global_load_async_to_lds_b128)
__device__ __forceinline__ void async_cp16(unsigned short* lds,
                                           const unsigned short* g) {
  __builtin_amdgcn_global_load_async_to_lds_b128(
      (__attribute__((address_space(1))) v4i_*)g,
      (__attribute__((address_space(3))) v4i_*)lds, 0, 0);
}
#else
__device__ __forceinline__ void async_cp16(unsigned short* lds,
                                           const unsigned short* g) {
  unsigned loff = (unsigned)(unsigned long long)lds;  // flat LDS low 32 bits
  asm volatile("global_load_async_to_lds_b128 %0, %1, off"
               :: "v"(loff), "v"(g) : "memory");
}
#endif

#if __has_builtin(__builtin_amdgcn_s_wait_asynccnt)
#define WAIT_ASYNC(n) __builtin_amdgcn_s_wait_asynccnt(n)
#else
#define WAIT_ASYNC(n) asm volatile("s_wait_asynccnt %0" :: "n"(n) : "memory")
#endif

// ----------------------------- fp32 -> bf16 --------------------------------
__global__ __launch_bounds__(256) void cvt_bf16(const float* __restrict__ in,
                                                unsigned short* __restrict__ out,
                                                int n4) {
  int i = blockIdx.x * blockDim.x + threadIdx.x;
  if (i < n4) {
    float4 f = ((const float4*)in)[i];
    ushort4 o;
    o.x = f2bf(f.x); o.y = f2bf(f.y); o.z = f2bf(f.z); o.w = f2bf(f.w);
    ((ushort4*)out)[i] = o;
  }
}

// ------------------------------- GEMM --------------------------------------
// C[N=8192, M=1024] = A[N,1024] @ W[M=1024, K=1024]^T  (bf16 in, fp32 acc)
// Block tile 128x64, 8 waves of 32x32; K-step 64, double-buffered async LDS.
// mode 0: bf16 row-major out. mode 1: fp32 + bias out. mode 2: bf16 out in
// per-head transposed layout [B*H*D=4096 rows][S=2048] (for flash V operand).
__global__ __launch_bounds__(256)
void gemm_bf16(const unsigned short* __restrict__ A,
               const unsigned short* __restrict__ W,
               void* __restrict__ outp,
               const float* __restrict__ bias,
               int mode) {
  __shared__ __align__(16) unsigned short As[2][128 * 72];  // 128 x 64, pad 8
  __shared__ __align__(16) unsigned short Bs[2][64 * 72];   // 64 x 64, pad 8

  const int t    = threadIdx.x;
  const int w    = t >> 5;
  const int lane = t & 31;
  const int half = lane >> 4;
  const int ln   = lane & 15;
  const int wm   = w & 3;   // 4-way M split (32 rows each)
  const int wn   = w >> 2;  // 2-way N split (32 cols each)
  const int m0   = blockIdx.x * 128;
  const int n0   = blockIdx.y * 64;

  // 6 async b128 instructions per wave per tile (A: 4, W: 2)
  auto issue = [&](int kt, int buf) {
    const int kb = kt * 64;
#pragma unroll
    for (int i = 0; i < 4; ++i) {
      int c = t + i * 256;          // 1024 chunks: 128 rows x 8
      int r = c >> 3, o = c & 7;
      async_cp16(&As[buf][r * 72 + o * 8],
                 A + (size_t)(m0 + r) * 1024 + kb + o * 8);
    }
#pragma unroll
    for (int i = 0; i < 2; ++i) {
      int c = t + i * 256;          // 512 chunks: 64 rows x 8
      int r = c >> 3, o = c & 7;
      async_cp16(&Bs[buf][r * 72 + o * 8],
                 W + (size_t)(n0 + r) * 1024 + kb + o * 8);
    }
  };

  issue(0, 0);
  v8f acc[2][2] = {};

  for (int kt = 0; kt < 16; ++kt) {
    const int cur = kt & 1;
    if (kt + 1 < 16) { issue(kt + 1, cur ^ 1); WAIT_ASYNC(6); }
    else             { WAIT_ASYNC(0); }
    __syncthreads();  // everyone's async chunks for tile kt have landed

#pragma unroll
    for (int ks = 0; ks < 2; ++ks) {
      FragB a[2], b[2];
#pragma unroll
      for (int mt = 0; mt < 2; ++mt) {  // A-operand layout
        const unsigned short* p = &As[cur][(wm * 32 + mt * 16 + ln) * 72 + ks * 32];
        a[mt].q[0] = *(const uint4*)(p + half * 8);
        a[mt].q[1] = *(const uint4*)(p + 16 + half * 8);
      }
#pragma unroll
      for (int nt = 0; nt < 2; ++nt) {  // B-operand layout
        const unsigned short* p = &Bs[cur][(wn * 32 + nt * 16 + ln) * 72 + ks * 32];
        b[nt].q[0] = *(const uint4*)(p + half * 8);
        b[nt].q[1] = *(const uint4*)(p + 16 + half * 8);
      }
#pragma unroll
      for (int mt = 0; mt < 2; ++mt)
#pragma unroll
        for (int nt = 0; nt < 2; ++nt)
          acc[mt][nt] = __builtin_amdgcn_wmma_f32_16x16x32_bf16(
              false, a[mt].v, false, b[nt].v, (short)0, acc[mt][nt],
              false, false);
    }
    __syncthreads();  // done reading buf `cur` before it is refilled
  }

#pragma unroll
  for (int mt = 0; mt < 2; ++mt)
#pragma unroll
    for (int nt = 0; nt < 2; ++nt)
#pragma unroll
      for (int r = 0; r < 8; ++r) {
        int row = m0 + wm * 32 + mt * 16 + r + 8 * half;  // C/D layout
        int col = n0 + wn * 32 + nt * 16 + ln;
        float v = acc[mt][nt][r];
        if (mode == 1) {
          ((float*)outp)[(size_t)row * 1024 + col] = v + bias[col];
        } else if (mode == 0) {
          ((unsigned short*)outp)[(size_t)row * 1024 + col] = f2bf(v);
        } else {
          // per-head transposed: vt[(b*1024 + col)][s], row = b*2048 + s
          ((unsigned short*)outp)[((size_t)(row >> 11) * 1024 + col) * 2048 +
                                  (row & 2047)] = f2bf(v);
        }
      }
}

// --------------------------- Flash attention -------------------------------
// One WG per (qtile=128 rows, head, batch); 8 waves x 16 query rows.
// Double-buffered async staging of 32-key K blocks and V^T blocks.
__global__ __launch_bounds__(256)
void flash_attn(const unsigned short* __restrict__ Qg,   // [8192,1024]
                const unsigned short* __restrict__ Kg,   // [8192,1024]
                const unsigned short* __restrict__ VtG,  // [4096,2048]
                unsigned short* __restrict__ Og) {       // [8192,1024]
  __shared__ __align__(16) unsigned short Ks[2][32 * 72];  // 32 keys x 64 d
  __shared__ __align__(16) unsigned short Vs[2][64 * 40];  // 64 d x 32 keys
  __shared__ __align__(16) unsigned short Ps[8][16 * 40];  // per-wave P tile

  const int t    = threadIdx.x;
  const int w    = t >> 5;
  const int lane = t & 31;
  const int half = lane >> 4;
  const int ln   = lane & 15;
  const int qt = blockIdx.x, h = blockIdx.y, bb = blockIdx.z;
  const int headOff = h * 64;
  const int rowB    = bb * 2048;
  const int q0      = qt * 128 + w * 16;

  // 2 async b128 instructions per wave per tile (K: 1, V^T: 1)
  auto issue = [&](int kt, int buf) {
    const int kb = kt * 32;
    {
      int r = t >> 3, c = t & 7;  // 256 chunks: 32 rows x 8
      async_cp16(&Ks[buf][r * 72 + c * 8],
                 Kg + (size_t)(rowB + kb + r) * 1024 + headOff + c * 8);
    }
    {
      int d = t >> 2, c = t & 3;  // 256 chunks: 64 rows x 4
      async_cp16(&Vs[buf][d * 40 + c * 8],
                 VtG + (size_t)(bb * 1024 + headOff + d) * 2048 + kb + c * 8);
    }
  };

  // Q fragments (A layout), registers for the whole sweep.
  FragB aq[2];
  {
    const unsigned short* qp = Qg + (size_t)(rowB + q0 + ln) * 1024 + headOff;
#pragma unroll
    for (int ks = 0; ks < 2; ++ks) {
      aq[ks].q[0] = *(const uint4*)(qp + ks * 32 + half * 8);
      aq[ks].q[1] = *(const uint4*)(qp + ks * 32 + 16 + half * 8);
    }
  }

  v8f o[4] = {};
  float m_r[8], l_r[8];
#pragma unroll
  for (int r = 0; r < 8; ++r) { m_r[r] = -1e30f; l_r[r] = 0.0f; }

  const float C2 = 0.125f * 1.44269504f;  // (1/sqrt(64)) * log2(e)

  issue(0, 0);

  for (int kt = 0; kt < 64; ++kt) {
    const int cur = kt & 1;
    if (kt + 1 < 64) { issue(kt + 1, cur ^ 1); WAIT_ASYNC(2); }
    else             { WAIT_ASYNC(0); }
    __syncthreads();

    // S = Q @ K^T : two 16x16 tiles over 32 keys
    v8f s0 = {}, s1 = {};
#pragma unroll
    for (int ks = 0; ks < 2; ++ks) {
      FragB bk0, bk1;
      const unsigned short* kp0 = &Ks[cur][(0 * 16 + ln) * 72 + ks * 32];
      const unsigned short* kp1 = &Ks[cur][(1 * 16 + ln) * 72 + ks * 32];
      bk0.q[0] = *(const uint4*)(kp0 + half * 8);
      bk0.q[1] = *(const uint4*)(kp0 + 16 + half * 8);
      bk1.q[0] = *(const uint4*)(kp1 + half * 8);
      bk1.q[1] = *(const uint4*)(kp1 + 16 + half * 8);
      s0 = __builtin_amdgcn_wmma_f32_16x16x32_bf16(false, aq[ks].v, false,
                                                   bk0.v, (short)0, s0,
                                                   false, false);
      s1 = __builtin_amdgcn_wmma_f32_16x16x32_bf16(false, aq[ks].v, false,
                                                   bk1.v, (short)0, s1,
                                                   false, false);
    }

    // online softmax (row = r + 8*half; 16-lane row reductions stay in-half)
#pragma unroll
    for (int r = 0; r < 8; ++r) {
      float mx = fmaxf(s0[r], s1[r]);
#pragma unroll
      for (int off = 1; off < 16; off <<= 1)
        mx = fmaxf(mx, __shfl_xor(mx, off, 32));
      float mnew  = fmaxf(m_r[r], mx);
      float alpha = exp2f((m_r[r] - mnew) * C2);
      float p0    = exp2f((s0[r] - mnew) * C2);
      float p1    = exp2f((s1[r] - mnew) * C2);
      float sum = p0 + p1;
#pragma unroll
      for (int off = 1; off < 16; off <<= 1)
        sum += __shfl_xor(sum, off, 32);
      l_r[r] = l_r[r] * alpha + sum;
      m_r[r] = mnew;
#pragma unroll
      for (int nt = 0; nt < 4; ++nt) o[nt][r] *= alpha;
      // C-layout -> LDS (for A-layout readback)
      int m = r + 8 * half;
      Ps[w][m * 40 + ln]      = f2bf(p0);
      Ps[w][m * 40 + 16 + ln] = f2bf(p1);
    }
    // DS ops of one wave execute in order at the LDS; just keep the compiler
    // from reordering the aliasing accesses.
    asm volatile("" ::: "memory");

    FragB ap;  // P as A operand
    {
      const unsigned short* pp = Ps[w] + ln * 40;
      ap.q[0] = *(const uint4*)(pp + half * 8);
      ap.q[1] = *(const uint4*)(pp + 16 + half * 8);
    }
    // O += P @ V : 4 dim-tiles, K = 32 keys
#pragma unroll
    for (int nt = 0; nt < 4; ++nt) {
      FragB bv;
      const unsigned short* vp = &Vs[cur][(nt * 16 + ln) * 40];
      bv.q[0] = *(const uint4*)(vp + half * 8);
      bv.q[1] = *(const uint4*)(vp + 16 + half * 8);
      o[nt] = __builtin_amdgcn_wmma_f32_16x16x32_bf16(false, ap.v, false,
                                                      bv.v, (short)0, o[nt],
                                                      false, false);
    }
    __syncthreads();  // done with buf `cur` before it is refilled
  }

  // normalize and emit bf16 attention output
#pragma unroll
  for (int r = 0; r < 8; ++r) {
    float inv = 1.0f / l_r[r];
    int row = q0 + r + 8 * half;
#pragma unroll
    for (int nt = 0; nt < 4; ++nt)
      Og[(size_t)(rowB + row) * 1024 + headOff + nt * 16 + ln] =
          f2bf(o[nt][r] * inv);
  }
}

// ------------------------------- launch ------------------------------------
extern "C" void kernel_launch(void* const* d_in, const int* in_sizes, int n_in,
                              void* d_out, int out_size, void* d_ws,
                              size_t ws_size, hipStream_t stream) {
  const float* x   = (const float*)d_in[0];  // [4,2048,1024]
  const float* W_q = (const float*)d_in[1];  // [1024,1024]
  const float* W_k = (const float*)d_in[2];
  const float* W_v = (const float*)d_in[3];
  const float* W_o = (const float*)d_in[4];
  const float* b_o = (const float*)d_in[5];  // [1024]
  float* out = (float*)d_out;                // [4,2048,1024]

  const size_t N = 8192;  // B*S
  const size_t E = 1024;
  char* ws = (char*)d_ws;
  size_t off = 0;
  unsigned short* xb  = (unsigned short*)(ws + off); off += N * E * 2;
  unsigned short* wqb = (unsigned short*)(ws + off); off += E * E * 2;
  unsigned short* wkb = (unsigned short*)(ws + off); off += E * E * 2;
  unsigned short* wvb = (unsigned short*)(ws + off); off += E * E * 2;
  unsigned short* wob = (unsigned short*)(ws + off); off += E * E * 2;
  unsigned short* qb  = (unsigned short*)(ws + off); off += N * E * 2;
  unsigned short* kb  = (unsigned short*)(ws + off); off += N * E * 2;
  unsigned short* vtb = (unsigned short*)(ws + off); off += N * E * 2;  // [4096,2048]
  unsigned short* ab  = (unsigned short*)(ws + off); off += N * E * 2;

  // 1) fp32 -> bf16 conversions
  {
    int n4 = (int)(N * E / 4);
    cvt_bf16<<<(n4 + 255) / 256, 256, 0, stream>>>(x, xb, n4);
    int w4 = (int)(E * E / 4);
    cvt_bf16<<<(w4 + 255) / 256, 256, 0, stream>>>(W_q, wqb, w4);
    cvt_bf16<<<(w4 + 255) / 256, 256, 0, stream>>>(W_k, wkb, w4);
    cvt_bf16<<<(w4 + 255) / 256, 256, 0, stream>>>(W_v, wvb, w4);
    cvt_bf16<<<(w4 + 255) / 256, 256, 0, stream>>>(W_o, wob, w4);
  }

  // 2) Q/K/V projections: [8192,1024] @ W^T  (V written per-head transposed)
  dim3 ggrid(64, 16);  // (8192/128, 1024/64)
  gemm_bf16<<<ggrid, 256, 0, stream>>>(xb, wqb, (void*)qb,  nullptr, 0);
  gemm_bf16<<<ggrid, 256, 0, stream>>>(xb, wkb, (void*)kb,  nullptr, 0);
  gemm_bf16<<<ggrid, 256, 0, stream>>>(xb, wvb, (void*)vtb, nullptr, 2);

  // 3) fused flash attention per (b, h, 128-query tile)
  flash_attn<<<dim3(16, 16, 4), 256, 0, stream>>>(qb, kb, vtb, ab);

  // 4) output projection + bias, fp32 out
  gemm_bf16<<<ggrid, 256, 0, stream>>>(ab, wob, (void*)out, b_o, 1);

  (void)in_sizes; (void)n_in; (void)out_size; (void)ws_size;
}
